// CFGDualSelfAttention_77412490543666
// MI455X (gfx1250) — matrix operand order repaired and checked by reading, not compiled, not run on hardware
//
#include <hip/hip_runtime.h>
#include <hip/hip_bf16.h>

// ---------------------------------------------------------------------------
// Types
// ---------------------------------------------------------------------------
typedef __bf16 bf16_t;
typedef bf16_t v16bf __attribute__((ext_vector_type(16)));
typedef bf16_t v2bf  __attribute__((ext_vector_type(2)));
typedef float  v8f   __attribute__((ext_vector_type(8)));
typedef float  vf4   __attribute__((ext_vector_type(4)));
typedef unsigned int u32x4 __attribute__((ext_vector_type(4)));
typedef unsigned int u32x2 __attribute__((ext_vector_type(2)));
typedef int v4i_vs __attribute__((vector_size(4 * sizeof(int)))); // matches builtin param

union Frag { u32x4 u[2]; v16bf v; };   // 16 bf16 = 8 VGPRs = one WMMA A/B operand

__device__ __forceinline__ unsigned short f2bf(float f) {
    unsigned u = __float_as_uint(f);
    u += 0x7FFFu + ((u >> 16) & 1u);          // round-to-nearest-even
    return (unsigned short)(u >> 16);
}
// Pack two fp32 -> packed bf16x2.  Prefer hardware v_cvt_pk_bf16_f32.
__device__ __forceinline__ unsigned pack2(float lo, float hi) {
#if defined(__AMDGCN__) && __has_builtin(__builtin_amdgcn_cvt_pk_bf16_f32)
    v2bf p = __builtin_amdgcn_cvt_pk_bf16_f32(lo, hi);
    return __builtin_bit_cast(unsigned, p);
#else
    return (unsigned)f2bf(lo) | ((unsigned)f2bf(hi) << 16);
#endif
}

#define WMMA_BF16(A, B, C) \
    __builtin_amdgcn_wmma_f32_16x16x32_bf16(false, (A), false, (B), (short)0, (C), false, false)

// Async copy to LDS (gfx1250 GLOBAL_LOAD_ASYNC_TO_LDS_B128), guarded.
#if defined(__AMDGCN__) && __has_builtin(__builtin_amdgcn_global_load_async_to_lds_b128)
#define HAVE_ASYNC_LDS 1
#define GLOBAL_AS __attribute__((address_space(1)))
#define LDS_AS    __attribute__((address_space(3)))
#else
#define HAVE_ASYNC_LDS 0
#endif

// ---------------------------------------------------------------------------
// Constants
// ---------------------------------------------------------------------------
#define DIM   5120
#define SEQ   2048
#define NH    40
#define HD    128
#define M_TOT 4096           // 2 * SEQ
#define QK_N  15360          // 3 * DIM

// ===========================================================================
// Kernel 1/4: tiled bf16 WMMA GEMM.  BM=BN=128, BK=32, 8 waves (wave32).
//   MODE 0: A = concat(hidden_cond, hidden_uncond) fp32, B = Wqkv,
//           epilogue splits -> Q fp32, K fp32, V bf16 (b,h,s,d)
//   MODE 1: A = attention output bf16, B = Wout, C = d_out fp32 (+bias)
// ===========================================================================
template <int MODE>
__global__ __launch_bounds__(256)
void gemm_bf16_kernel(const float* __restrict__ Acond,
                      const float* __restrict__ Auncond,
                      const unsigned short* __restrict__ Abf,
                      const float* __restrict__ Bm,
                      const float* __restrict__ bias,
                      float* __restrict__ Qout,
                      float* __restrict__ Kout,
                      unsigned short* __restrict__ Vout,
                      float* __restrict__ Cout,
                      int N)
{
    __shared__ __align__(16) unsigned short smA[2][128 * 40]; // [row][k], stride 40 (pad)
    __shared__ __align__(16) unsigned short smB[2][128 * 40]; // [n][k] transposed, stride 40

    const int t    = threadIdx.x;
    const int lane = t & 31, lr = lane & 15, lh = lane >> 4;
    const int w    = t >> 5, wm = w & 3, wn = w >> 2;
    const int m0   = blockIdx.y * 128;
    const int n0   = blockIdx.x * 128;

    v8f acc[2][4];
    const v8f z8 = {0.f, 0.f, 0.f, 0.f, 0.f, 0.f, 0.f, 0.f};
#pragma unroll
    for (int i = 0; i < 2; ++i)
#pragma unroll
        for (int j = 0; j < 4; ++j) acc[i][j] = z8;

    auto stageA = [&](int buf, int k0) {
        const int r = t >> 1, cB = (t & 1) * 16;
        const int gr = m0 + r;
        if (MODE == 0) {
            const float* src = (gr < SEQ) ? (Acond + (size_t)gr * DIM)
                                          : (Auncond + (size_t)(gr - SEQ) * DIM);
            src += k0 + cB;
            vf4 f[4];
#pragma unroll
            for (int jj = 0; jj < 4; ++jj) f[jj] = *(const vf4*)(src + jj * 4);
            u32x4 lo = {pack2(f[0][0], f[0][1]), pack2(f[0][2], f[0][3]),
                        pack2(f[1][0], f[1][1]), pack2(f[1][2], f[1][3])};
            u32x4 hi = {pack2(f[2][0], f[2][1]), pack2(f[2][2], f[2][3]),
                        pack2(f[3][0], f[3][1]), pack2(f[3][2], f[3][3])};
            *(u32x4*)&smA[buf][r * 40 + cB]     = lo;
            *(u32x4*)&smA[buf][r * 40 + cB + 8] = hi;
        } else {
            const unsigned short* src = Abf + (size_t)gr * DIM + k0 + cB;
            *(u32x4*)&smA[buf][r * 40 + cB]     = *(const u32x4*)src;
            *(u32x4*)&smA[buf][r * 40 + cB + 8] = *(const u32x4*)(src + 8);
        }
    };

    auto stageB = [&](int buf, int k0) {
        const int nB = (t & 31) * 4, kB = (t >> 5) * 4;
        const float* src = Bm + (size_t)(k0 + kB) * N + (n0 + nB);
        vf4 f0 = *(const vf4*)(src);
        vf4 f1 = *(const vf4*)(src + (size_t)N);
        vf4 f2 = *(const vf4*)(src + 2 * (size_t)N);
        vf4 f3 = *(const vf4*)(src + 3 * (size_t)N);
#pragma unroll
        for (int j = 0; j < 4; ++j) {          // register 4x4 transpose -> k-contig
            u32x2 o2 = {pack2(f0[j], f1[j]), pack2(f2[j], f3[j])};
            *(u32x2*)&smB[buf][(nB + j) * 40 + kB] = o2;
        }
    };

    auto computeTile = [&](int buf) {
        Frag af[2], bf[4];
#pragma unroll
        for (int i = 0; i < 2; ++i) {
            const int row = wm * 32 + i * 16 + lr;
            af[i].u[0] = *(const u32x4*)&smA[buf][row * 40 + lh * 8];
            af[i].u[1] = *(const u32x4*)&smA[buf][row * 40 + 16 + lh * 8];
        }
#pragma unroll
        for (int j = 0; j < 4; ++j) {
            const int n = wn * 64 + j * 16 + lr;
            bf[j].u[0] = *(const u32x4*)&smB[buf][n * 40 + lh * 8];
            bf[j].u[1] = *(const u32x4*)&smB[buf][n * 40 + 16 + lh * 8];
        }
#pragma unroll
        for (int i = 0; i < 2; ++i)
#pragma unroll
            for (int j = 0; j < 4; ++j)
                acc[i][j] = WMMA_BF16(af[i].v, bf[j].v, acc[i][j]);
    };

    stageA(0, 0);
    stageB(0, 0);
    const int NT = DIM / 32;   // 160
    for (int kt = 0; kt < NT; ++kt) {
        __syncthreads();
        const int cur = kt & 1;
        if (kt + 1 < NT) { stageA(cur ^ 1, (kt + 1) * 32); stageB(cur ^ 1, (kt + 1) * 32); }
        computeTile(cur);
    }

    // Epilogue.  C layout: vgpr r at lane -> row (r + 8*lh), col lr of the 16x16 tile.
#pragma unroll
    for (int i = 0; i < 2; ++i)
#pragma unroll
        for (int j = 0; j < 4; ++j)
#pragma unroll
            for (int r = 0; r < 8; ++r) {
                const int m = m0 + wm * 32 + i * 16 + r + 8 * lh;
                const int n = n0 + wn * 64 + j * 16 + lr;
                const float val = acc[i][j][r] + bias[n];
                if (MODE == 0) {
                    if (n < DIM) {
                        Qout[(size_t)m * DIM + n] = val;
                    } else if (n < 2 * DIM) {
                        Kout[(size_t)m * DIM + (n - DIM)] = val;
                    } else {
                        const int c = n - 2 * DIM;
                        const int hh = c >> 7, d = c & 127;
                        const int b = m >> 11, s = m & 2047;
                        Vout[(((size_t)(b * NH + hh) * SEQ) + s) * HD + d] = f2bf(val);
                    }
                } else {
                    Cout[(size_t)m * N + n] = val;
                }
            }
}

// ===========================================================================
// Kernel 2: fused RMSNorm (over DIM) + RoPE + bf16 pack, one block per token.
// Folds HD^-0.5 into Q.  Output layout (b,h,s,d) for attention.
// ===========================================================================
__global__ __launch_bounds__(256)
void norm_rope_kernel(const float* __restrict__ Qr, const float* __restrict__ Kr,
                      const float* __restrict__ cosT, const float* __restrict__ sinT,
                      const float* __restrict__ wq, const float* __restrict__ wk,
                      unsigned short* __restrict__ Qb, unsigned short* __restrict__ Kb)
{
    __shared__ float sq[DIM];
    __shared__ float sk[DIM];
    __shared__ float red[2][8];

    const int t   = threadIdx.x;
    const int tok = blockIdx.x;
    const int b   = tok >> 11, s = tok & 2047;
    const float* qsrc = Qr + (size_t)tok * DIM;
    const float* ksrc = Kr + (size_t)tok * DIM;

    float aq = 0.f, ak = 0.f;
    for (int i = t; i < DIM; i += 256) {
        const float q = qsrc[i], k = ksrc[i];
        sq[i] = q; sk[i] = k;
        aq += q * q; ak += k * k;
    }
#pragma unroll
    for (int off = 16; off >= 1; off >>= 1) {
        aq += __shfl_xor(aq, off, 32);
        ak += __shfl_xor(ak, off, 32);
    }
    if ((t & 31) == 0) { red[0][t >> 5] = aq; red[1][t >> 5] = ak; }
    __syncthreads();
    float tq = 0.f, tk = 0.f;
#pragma unroll
    for (int wv = 0; wv < 8; ++wv) { tq += red[0][wv]; tk += red[1][wv]; }
    const float rq = rsqrtf(tq * (1.f / DIM) + 1e-6f);
    const float rk = rsqrtf(tk * (1.f / DIM) + 1e-6f);
    const float qscale = 0.08838834764831845f;   // HD^-0.5

    for (int i = t; i < DIM; i += 256) {
        const int d = i & 127, hh = i >> 7;
        const float c  = cosT[s * HD + d];
        const float sn = sinT[s * HD + d];
        const int ip   = hh * HD + ((d < 64) ? d + 64 : d - 64);
        const float sgn = (d < 64) ? -1.f : 1.f;
        const float qn = sq[i] * rq * wq[i], qp = sq[ip] * rq * wq[ip];
        const float kn = sk[i] * rk * wk[i], kp = sk[ip] * rk * wk[ip];
        const size_t o = (((size_t)(b * NH + hh) * SEQ) + s) * HD + d;
        Qb[o] = f2bf((qn * c + sgn * qp * sn) * qscale);
        Kb[o] = f2bf(kn * c + sgn * kp * sn);
    }
}

// ===========================================================================
// Kernel 3: flash attention.  Block = 128 Q rows of one (b,h); 8 waves, each
// wave owns 16 rows.  Q fragments live in registers; K tile staged to LDS
// (async-to-LDS path when available); V transposed into LDS for the P.V WMMA.
// ===========================================================================
__global__ __launch_bounds__(256)
void attn_kernel(const unsigned short* __restrict__ Qb,
                 const unsigned short* __restrict__ Kb,
                 const unsigned short* __restrict__ Vb,
                 unsigned short* __restrict__ Ob)
{
    __shared__ __align__(16) unsigned short smK[64 * 136];    // [s'][d], pad 8
    __shared__ __align__(16) unsigned short smVt[128 * 72];   // [d][s'], pad 8
    __shared__ __align__(16) unsigned short smP[8 * 16 * 72]; // per-wave P, [row][s']

    const int t = threadIdx.x;
    const int lane = t & 31, lr = lane & 15, lh = lane >> 4;
    const int w = t >> 5;
    const int qt = blockIdx.x, bh = blockIdx.y;
    const int b = bh / NH, h = bh % NH;

    const unsigned short* Qh = Qb + (size_t)bh * SEQ * HD;
    const unsigned short* Kh = Kb + (size_t)bh * SEQ * HD;
    const unsigned short* Vh = Vb + (size_t)bh * SEQ * HD;

    // Q fragments (A-operand): wave owns rows [qrow0, qrow0+16)
    const int qrow0 = qt * 128 + w * 16;
    Frag qf[4];
#pragma unroll
    for (int kt = 0; kt < 4; ++kt) {
        const unsigned short* qp = Qh + (size_t)(qrow0 + lr) * HD + kt * 32;
        qf[kt].u[0] = *(const u32x4*)(qp + lh * 8);
        qf[kt].u[1] = *(const u32x4*)(qp + 16 + lh * 8);
    }

    const v8f z8 = {0.f, 0.f, 0.f, 0.f, 0.f, 0.f, 0.f, 0.f};
    v8f o[8];
#pragma unroll
    for (int dt = 0; dt < 8; ++dt) o[dt] = z8;
    float mrow[8], lrow[8];
#pragma unroll
    for (int r = 0; r < 8; ++r) { mrow[r] = -1e30f; lrow[r] = 0.f; }

    for (int kv = 0; kv < SEQ / 64; ++kv) {
        const int s0 = kv * 64;
        __syncthreads();   // protect LDS tiles from previous iteration

        // ---- stage K tile 64x128 bf16 (natural layout == B^T for QK^T) ----
#if HAVE_ASYNC_LDS
#pragma unroll
        for (int j = 0; j < 4; ++j) {
            const int chunk = j * 256 + t;
            const int r = chunk >> 4, c = chunk & 15;
            const unsigned short* src = Kh + (size_t)(s0 + r) * HD + c * 8;
            unsigned short* dst = &smK[r * 136 + c * 8];
            __builtin_amdgcn_global_load_async_to_lds_b128(
                (GLOBAL_AS v4i_vs*)src, (LDS_AS v4i_vs*)dst, 0, 0);
        }
#else
#pragma unroll
        for (int j = 0; j < 4; ++j) {
            const int chunk = j * 256 + t;
            const int r = chunk >> 4, c = chunk & 15;
            *(u32x4*)&smK[r * 136 + c * 8] =
                *(const u32x4*)(Kh + (size_t)(s0 + r) * HD + c * 8);
        }
#endif
        // ---- stage V tile transposed: smVt[d][s'] ----
#pragma unroll
        for (int i = 0; i < 2; ++i) {
            const int blk = i * 256 + t;
            const int d = (blk & 31) * 4, sp = (blk >> 5) * 4;
            unsigned short e[4][4];
#pragma unroll
            for (int r2 = 0; r2 < 4; ++r2) {
                u32x2 v2 = *(const u32x2*)(Vh + (size_t)(s0 + sp + r2) * HD + d);
                e[r2][0] = (unsigned short)(v2[0] & 0xffffu);
                e[r2][1] = (unsigned short)(v2[0] >> 16);
                e[r2][2] = (unsigned short)(v2[1] & 0xffffu);
                e[r2][3] = (unsigned short)(v2[1] >> 16);
            }
#pragma unroll
            for (int jj = 0; jj < 4; ++jj) {
                u32x2 o2 = {(unsigned)e[0][jj] | ((unsigned)e[1][jj] << 16),
                            (unsigned)e[2][jj] | ((unsigned)e[3][jj] << 16)};
                *(u32x2*)&smVt[(d + jj) * 72 + sp] = o2;
            }
        }
#if HAVE_ASYNC_LDS
#if __has_builtin(__builtin_amdgcn_s_wait_asynccnt)
        __builtin_amdgcn_s_wait_asynccnt(0);
#else
        asm volatile("s_wait_asynccnt 0" ::: "memory");
#endif
#endif
        __syncthreads();

        // ---- S = Q @ K^T : 16x64 per wave ----
        v8f sacc[4];
#pragma unroll
        for (int j = 0; j < 4; ++j) sacc[j] = z8;
#pragma unroll
        for (int j = 0; j < 4; ++j)
#pragma unroll
            for (int kt = 0; kt < 4; ++kt) {
                Frag bf;
                const unsigned short* kp = &smK[(j * 16 + lr) * 136 + kt * 32 + lh * 8];
                bf.u[0] = *(const u32x4*)kp;
                bf.u[1] = *(const u32x4*)(kp + 16);
                sacc[j] = WMMA_BF16(qf[kt].v, bf.v, sacc[j]);
            }

        // ---- online softmax (rows split across lane halves; xor 1..8) ----
        float newm[8], alpha[8], psum[8];
#pragma unroll
        for (int r = 0; r < 8; ++r) {
            float mx = fmaxf(fmaxf(sacc[0][r], sacc[1][r]), fmaxf(sacc[2][r], sacc[3][r]));
#pragma unroll
            for (int off = 8; off >= 1; off >>= 1) mx = fmaxf(mx, __shfl_xor(mx, off, 32));
            newm[r]  = fmaxf(mrow[r], mx);
            alpha[r] = __expf(mrow[r] - newm[r]);
            mrow[r]  = newm[r];
            psum[r]  = 0.f;
        }
#pragma unroll
        for (int j = 0; j < 4; ++j)
#pragma unroll
            for (int r = 0; r < 8; ++r) {
                const float p = __expf(sacc[j][r] - newm[r]);
                psum[r] += p;
                smP[(w * 16 + r + 8 * lh) * 72 + j * 16 + lr] = f2bf(p);
            }
#pragma unroll
        for (int r = 0; r < 8; ++r) {
            float sps = psum[r];
#pragma unroll
            for (int off = 8; off >= 1; off >>= 1) sps += __shfl_xor(sps, off, 32);
            lrow[r] = lrow[r] * alpha[r] + sps;
        }
#pragma unroll
        for (int dt = 0; dt < 8; ++dt)
#pragma unroll
            for (int r = 0; r < 8; ++r) o[dt][r] *= alpha[r];

        // ---- O += P @ V ----
#pragma unroll
        for (int dt = 0; dt < 8; ++dt)
#pragma unroll
            for (int k2 = 0; k2 < 2; ++k2) {
                Frag pa, vbf;
                const unsigned short* pp = &smP[(w * 16 + lr) * 72 + k2 * 32 + lh * 8];
                pa.u[0] = *(const u32x4*)pp;
                pa.u[1] = *(const u32x4*)(pp + 16);
                const unsigned short* vp = &smVt[(dt * 16 + lr) * 72 + k2 * 32 + lh * 8];
                vbf.u[0] = *(const u32x4*)vp;
                vbf.u[1] = *(const u32x4*)(vp + 16);
                o[dt] = WMMA_BF16(pa.v, vbf.v, o[dt]);
            }
    }

    // ---- epilogue: O /= l, write bf16 at (b, s, h*HD+d) for the out GEMM ----
#pragma unroll
    for (int r = 0; r < 8; ++r) {
        const float inv = 1.f / lrow[r];
        const int m = b * SEQ + qt * 128 + w * 16 + r + 8 * lh;
#pragma unroll
        for (int dt = 0; dt < 8; ++dt) {
            const int col = h * HD + dt * 16 + lr;
            Ob[(size_t)m * DIM + col] = f2bf(o[dt][r] * inv);
        }
    }
}

// ===========================================================================
// Launch
// ===========================================================================
extern "C" void kernel_launch(void* const* d_in, const int* in_sizes, int n_in,
                              void* d_out, int out_size, void* d_ws, size_t ws_size,
                              hipStream_t stream) {
    const float* hidden_cond   = (const float*)d_in[0];
    const float* hidden_uncond = (const float*)d_in[1];
    const float* cosT          = (const float*)d_in[2];
    const float* sinT          = (const float*)d_in[3];
    const float* Wqkv          = (const float*)d_in[4];
    const float* bqkv          = (const float*)d_in[5];
    const float* wq_norm       = (const float*)d_in[6];
    const float* wk_norm       = (const float*)d_in[7];
    const float* Wout          = (const float*)d_in[8];
    const float* bout          = (const float*)d_in[9];

    char* ws = (char*)d_ws;
    const size_t f32_mat = (size_t)M_TOT * DIM * sizeof(float);          // 84 MB
    const size_t bf_mat  = (size_t)M_TOT * DIM * sizeof(unsigned short); // 42 MB
    float* Qr = (float*)ws;                 ws += f32_mat;
    float* Kr = (float*)ws;                 ws += f32_mat;
    unsigned short* Qb = (unsigned short*)ws; ws += bf_mat;
    unsigned short* Kb = (unsigned short*)ws; ws += bf_mat;
    unsigned short* Vb = (unsigned short*)ws; ws += bf_mat;
    unsigned short* Ob = (unsigned short*)Qr;   // Qr dead after norm_rope; reuse

    // 1) QKV projection: 4096 x 15360, K=5120
    gemm_bf16_kernel<0><<<dim3(QK_N / 128, M_TOT / 128), 256, 0, stream>>>(
        hidden_cond, hidden_uncond, nullptr, Wqkv, bqkv, Qr, Kr, Vb, nullptr, QK_N);

    // 2) RMSNorm + RoPE, one block per token
    norm_rope_kernel<<<dim3(M_TOT), 256, 0, stream>>>(
        Qr, Kr, cosT, sinT, wq_norm, wk_norm, Qb, Kb);

    // 3) Flash attention: 16 q-tiles x 80 (b,h)
    attn_kernel<<<dim3(SEQ / 128, 2 * NH), 256, 0, stream>>>(Qb, Kb, Vb, Ob);

    // 4) Output projection: 4096 x 5120, K=5120 -> d_out (fp32)
    gemm_bf16_kernel<1><<<dim3(DIM / 128, M_TOT / 128), 256, 0, stream>>>(
        nullptr, nullptr, Ob, Wout, bout, nullptr, nullptr, nullptr, (float*)d_out, DIM);
}
